// DilatedAttention_83253646066154
// MI455X (gfx1250) — matrix operand order repaired and checked by reading, not compiled, not run on hardware
//
#include <hip/hip_runtime.h>
#include <cstdint>
#include <cstddef>

typedef __attribute__((ext_vector_type(16))) _Float16 v16h;
typedef __attribute__((ext_vector_type(8)))  float    v8f;

#define B_SZ   2
#define L_SZ   2048
#define D_SZ   1024
#define H_SZ   16
#define DH_SZ  64
#define HID_SZ 4096
#define DILC   2
#define WINC   256
#define EPSC   1e-5f

// ---------------------------------------------------------------------------
// CDNA5 async global->LDS DMA helpers (ASYNCcnt-tracked, no VGPR staging)
// ---------------------------------------------------------------------------
__device__ __forceinline__ uint32_t lds_off32(const void* p) {
  // generic pointer to LDS = {shared aperture, offset[31:0]}; LDS addr = low 32b
  return (uint32_t)(uintptr_t)p;
}
__device__ __forceinline__ void async_load_b128(uint32_t lds_dst,
                                                const void* gaddr) {
  asm volatile("global_load_async_to_lds_b128 %0, %1, off"
               :
               : "v"(lds_dst), "v"(gaddr)
               : "memory");
}
__device__ __forceinline__ void wait_async0() {
  asm volatile("s_wait_asynccnt 0x0" ::: "memory");
}

// ---------------------------------------------------------------------------
// f32 -> f16 weight conversion (grid-stride)
// ---------------------------------------------------------------------------
__global__ void cvt_f32_to_f16(const float* __restrict__ in,
                               _Float16* __restrict__ out, int n) {
  int i = blockIdx.x * blockDim.x + threadIdx.x;
  const int stride = gridDim.x * blockDim.x;
  for (; i < n; i += stride) out[i] = (_Float16)in[i];
}

// ---------------------------------------------------------------------------
// LayerNorm over D=1024, one block per row, f16 output for WMMA consumption
// ---------------------------------------------------------------------------
__global__ __launch_bounds__(256) void layernorm_to_f16(
    const float* __restrict__ x, const float* __restrict__ g,
    const float* __restrict__ bta, _Float16* __restrict__ out) {
  __shared__ float red[256];
  const int row = blockIdx.x;
  const float* xr = x + (size_t)row * D_SZ;
  float v[4];
  float s = 0.f;
#pragma unroll
  for (int t = 0; t < 4; ++t) { v[t] = xr[threadIdx.x + t * 256]; s += v[t]; }
  red[threadIdx.x] = s;
  __syncthreads();
  for (int off = 128; off > 0; off >>= 1) {
    if (threadIdx.x < off) red[threadIdx.x] += red[threadIdx.x + off];
    __syncthreads();
  }
  const float mean = red[0] * (1.f / D_SZ);
  __syncthreads();
  float s2 = 0.f;
#pragma unroll
  for (int t = 0; t < 4; ++t) { const float d = v[t] - mean; s2 += d * d; }
  red[threadIdx.x] = s2;
  __syncthreads();
  for (int off = 128; off > 0; off >>= 1) {
    if (threadIdx.x < off) red[threadIdx.x] += red[threadIdx.x + off];
    __syncthreads();
  }
  const float rstd = rsqrtf(red[0] * (1.f / D_SZ) + EPSC);
  _Float16* orow = out + (size_t)row * D_SZ;
#pragma unroll
  for (int t = 0; t < 4; ++t) {
    const int c = threadIdx.x + t * 256;
    orow[c] = (_Float16)((v[t] - mean) * rstd * g[c] + bta[c]);
  }
}

// ---------------------------------------------------------------------------
// Tiled WMMA GEMM:  C[M,N] = act(A[M,K] @ W[N,K]^T + bias) (+ res)
// 256 threads = 8 waves; block tile 128x128; K-step 32.
// Double-buffered LDS filled by GLOBAL_LOAD_ASYNC_TO_LDS_B128; one barrier
// per K-slice; async loads for slice k+1 overlap the 8 WMMAs of slice k.
// Wave grid 2(M)x4(N); each wave owns a 64x32 region = 4x2 16x16 subtiles.
// ---------------------------------------------------------------------------
union HFrag { uint32_t u[8]; v16h h; };

template <int ACT, bool RES, bool OUT16>
__global__ __launch_bounds__(256) void gemm_wmma_f16(
    const _Float16* __restrict__ A,   // [M,K] row-major
    const _Float16* __restrict__ W,   // [N,K] row-major
    const float* __restrict__ bias,   // [N]
    const float* __restrict__ res,    // [M,N] or unused
    void* __restrict__ outp,          // f16 or f32 [M,N]
    int M, int N, int K) {
  constexpr int LDSS = 40;  // halves per LDS row (80B: 16B-aligned, conflict-free)
  __shared__ __align__(16) _Float16 As[2][128 * LDSS];
  __shared__ __align__(16) _Float16 Bs[2][128 * LDSS];

  const int tid   = threadIdx.x;
  const int wave  = tid >> 5;
  const int lane  = tid & 31;
  const int lo    = lane & 15;
  const int hi    = lane >> 4;
  const int waveM = wave & 1;   // 0..1 -> 64 rows
  const int waveN = wave >> 1;  // 0..3 -> 32 cols

  const int m0 = blockIdx.y * 128;
  const int n0 = blockIdx.x * 128;

  // cooperative tile load mapping: 2 threads per row, one b128 (8 halves) each at half 0 or 16
  const int lrow = tid >> 1;
  const int lseg = (tid & 1) * 16;

  const _Float16* Aglob = A + (size_t)(m0 + lrow) * K + lseg;
  const _Float16* Wglob = W + (size_t)(n0 + lrow) * K + lseg;
  const uint32_t ldsA[2] = {lds_off32(&As[0][lrow * LDSS + lseg]),
                            lds_off32(&As[1][lrow * LDSS + lseg])};
  const uint32_t ldsB[2] = {lds_off32(&Bs[0][lrow * LDSS + lseg]),
                            lds_off32(&Bs[1][lrow * LDSS + lseg])};

  v8f acc[4][2] = {};

  // prologue: kick off DMA for slice 0
  async_load_b128(ldsA[0], Aglob);
  async_load_b128(ldsB[0], Wglob);

  int cur = 0;
  for (int k0 = 0; k0 < K; k0 += 32) {
    wait_async0();      // this wave's DMA into buf[cur] has landed
    __syncthreads();    // => every wave's DMA landed; buf[cur^1] reads done
    if (k0 + 32 < K) {  // start DMA for the next slice into the spare buffer
      async_load_b128(ldsA[cur ^ 1], Aglob + k0 + 32);
      async_load_b128(ldsB[cur ^ 1], Wglob + k0 + 32);
    }

    // A fragment (16x32, M x K): lane(lo)=row; hi selects K-half pattern.
    HFrag af[4], bf[2];
#pragma unroll
    for (int a = 0; a < 4; ++a) {
      const int rbase = (waveM * 64 + a * 16 + lo) * LDSS;
#pragma unroll
      for (int p = 0; p < 8; ++p) {
        const int kA = ((p < 4) ? 2 * p : 8 + 2 * p) + 8 * hi;
        af[a].u[p] = *(const uint32_t*)&As[cur][rbase + kA];
      }
    }
    // B fragment (32x16, K x N): lane(lo)=col n; same K pattern as A (table order).
#pragma unroll
    for (int bq = 0; bq < 2; ++bq) {
      const int rbase = (waveN * 32 + bq * 16 + lo) * LDSS;
#pragma unroll
      for (int p = 0; p < 8; ++p) {
        const int kB = ((p < 4) ? 2 * p : 8 + 2 * p) + 8 * hi;
        bf[bq].u[p] = *(const uint32_t*)&Bs[cur][rbase + kB];
      }
    }
#pragma unroll
    for (int a = 0; a < 4; ++a)
#pragma unroll
      for (int bq = 0; bq < 2; ++bq)
        acc[a][bq] = __builtin_amdgcn_wmma_f32_16x16x32_f16(
            false, af[a].h, false, bf[bq].h, (short)0, acc[a][bq], false, false);
    cur ^= 1;
  }

  // Epilogue: C/D layout -> m = r + 8*hi, n = lo within each 16x16 subtile.
  float* outF = (float*)outp;
  _Float16* outH = (_Float16*)outp;
#pragma unroll
  for (int a = 0; a < 4; ++a) {
    const int mloc = m0 + waveM * 64 + a * 16 + hi * 8;
#pragma unroll
    for (int bq = 0; bq < 2; ++bq) {
      const int n = n0 + waveN * 32 + bq * 16 + lo;
      const float bv = bias[n];
#pragma unroll
      for (int r = 0; r < 8; ++r) {
        const size_t idx = (size_t)(mloc + r) * N + n;
        float v = acc[a][bq][r] + bv;
        if (ACT == 1) v = 0.5f * v * (1.f + erff(v * 0.70710678118654752f));
        if (RES) v += res[idx];
        if (OUT16) outH[idx] = (_Float16)v;
        else       outF[idx] = v;
      }
    }
  }
}

// ---------------------------------------------------------------------------
// Dilated causal windowed attention: allowed j = i - 2t, t=0..128, j>=0.
// One thread per (b,h,i); one-pass online softmax, f32 accumulation.
// KV working set (~16 MB f16) is L2-resident (192 MB L2).
// ---------------------------------------------------------------------------
__global__ __launch_bounds__(128) void dilated_attn(
    const _Float16* __restrict__ qkv,  // [(B*L), 3D]; q|k|v split on last dim
    _Float16* __restrict__ y) {        // [(B*L), D]
  const int b = blockIdx.z, h = blockIdx.y;
  const int i = blockIdx.x * 128 + threadIdx.x;
  const _Float16* qr = qkv + (size_t)(b * L_SZ + i) * (3 * D_SZ) + h * DH_SZ;

  float q[DH_SZ];
#pragma unroll
  for (int d = 0; d < DH_SZ; ++d) q[d] = (float)qr[d];

  float m = -3.4e38f, l = 0.f;
  float acc[DH_SZ] = {};
  const float scale = 0.125f;    // 1/sqrt(DH)
  const int tmax = WINC / DILC;  // 128 -> up to 129 allowed keys

  for (int t = 0; t <= tmax; ++t) {
    const int j = i - t * DILC;
    if (j < 0) break;
    const _Float16* kr =
        qkv + (size_t)(b * L_SZ + j) * (3 * D_SZ) + D_SZ + h * DH_SZ;
    const _Float16* vr = kr + D_SZ;
    float s = 0.f;
#pragma unroll
    for (int d = 0; d < DH_SZ; ++d) s += q[d] * (float)kr[d];
    s *= scale;
    const float mn = fmaxf(m, s);
    const float corr = __expf(m - mn);  // first iter: exp(-inf)=0
    const float p = __expf(s - mn);
    l = l * corr + p;
#pragma unroll
    for (int d = 0; d < DH_SZ; ++d) acc[d] = acc[d] * corr + p * (float)vr[d];
    m = mn;
  }
  const float inv = 1.f / l;  // t=0 (j=i) always valid -> l > 0
  _Float16* yr = y + (size_t)(b * L_SZ + i) * D_SZ + h * DH_SZ;
#pragma unroll
  for (int d = 0; d < DH_SZ; ++d) yr[d] = (_Float16)(acc[d] * inv);
}

// ---------------------------------------------------------------------------
// Host orchestration
// ---------------------------------------------------------------------------
extern "C" void kernel_launch(void* const* d_in, const int* in_sizes, int n_in,
                              void* d_out, int out_size, void* d_ws,
                              size_t ws_size, hipStream_t stream) {
  (void)in_sizes; (void)n_in; (void)out_size; (void)ws_size;
  const float* x    = (const float*)d_in[0];
  const float* ln1g = (const float*)d_in[1];
  const float* ln1b = (const float*)d_in[2];
  const float* qkvw = (const float*)d_in[3];
  const float* qkvb = (const float*)d_in[4];
  const float* outw = (const float*)d_in[5];
  const float* outb = (const float*)d_in[6];
  const float* ln2g = (const float*)d_in[7];
  const float* ln2b = (const float*)d_in[8];
  const float* w1   = (const float*)d_in[9];
  const float* b1   = (const float*)d_in[10];
  const float* w2   = (const float*)d_in[11];
  const float* b2   = (const float*)d_in[12];
  float* out = (float*)d_out;

  const int M = B_SZ * L_SZ;  // 4096
  char* ws = (char*)d_ws;
  size_t off = 0;
  auto alloc = [&](size_t bytes) -> void* {
    void* p = ws + off;
    off += (bytes + 255) & ~(size_t)255;
    return p;
  };
  _Float16* qkvw16 = (_Float16*)alloc((size_t)3 * D_SZ * D_SZ * 2);
  _Float16* outw16 = (_Float16*)alloc((size_t)D_SZ * D_SZ * 2);
  _Float16* w1_16  = (_Float16*)alloc((size_t)HID_SZ * D_SZ * 2);
  _Float16* w2_16  = (_Float16*)alloc((size_t)D_SZ * HID_SZ * 2);
  _Float16* h16    = (_Float16*)alloc((size_t)M * D_SZ * 2);
  _Float16* qkv16  = (_Float16*)alloc((size_t)M * 3 * D_SZ * 2);
  _Float16* y16    = (_Float16*)alloc((size_t)M * D_SZ * 2);  // adjacent to qkv16
  float*    x1     = (float*)alloc((size_t)M * D_SZ * 4);
  // reuse: hid (4096x4096 f16 = 32MB) aliases qkv16+y16 (dead by FFN1);
  //        h2 aliases h1 (dead after QKV GEMM).
  _Float16* hid16 = qkv16;
  _Float16* h2_16 = h16;

  cvt_f32_to_f16<<<512, 256, 0, stream>>>(qkvw, qkvw16, 3 * D_SZ * D_SZ);
  cvt_f32_to_f16<<<512, 256, 0, stream>>>(outw, outw16, D_SZ * D_SZ);
  cvt_f32_to_f16<<<512, 256, 0, stream>>>(w1, w1_16, HID_SZ * D_SZ);
  cvt_f32_to_f16<<<512, 256, 0, stream>>>(w2, w2_16, D_SZ * HID_SZ);

  // h = LN1(x)
  layernorm_to_f16<<<M, 256, 0, stream>>>(x, ln1g, ln1b, h16);
  // qkv = h @ qkv_w^T + qkv_b        (f16 out)
  gemm_wmma_f16<0, false, true><<<dim3(3 * D_SZ / 128, M / 128), 256, 0, stream>>>(
      h16, qkvw16, qkvb, nullptr, qkv16, M, 3 * D_SZ, D_SZ);
  // y = dilated attention(q,k,v)     (f16 out)
  dilated_attn<<<dim3(L_SZ / 128, H_SZ, B_SZ), 128, 0, stream>>>(qkv16, y16);
  // x1 = y @ out_w^T + out_b + x     (f32 out, residual)
  gemm_wmma_f16<0, true, false><<<dim3(D_SZ / 128, M / 128), 256, 0, stream>>>(
      y16, outw16, outb, x, x1, M, D_SZ, D_SZ);
  // h2 = LN2(x1)
  layernorm_to_f16<<<M, 256, 0, stream>>>(x1, ln2g, ln2b, h2_16);
  // hid = gelu(h2 @ w1^T + b1)       (f16 out)
  gemm_wmma_f16<1, false, true><<<dim3(HID_SZ / 128, M / 128), 256, 0, stream>>>(
      h2_16, w1_16, b1, nullptr, hid16, M, HID_SZ, D_SZ);
  // out = hid @ w2^T + b2 + x1       (f32 out, residual)
  gemm_wmma_f16<0, true, false><<<dim3(D_SZ / 128, M / 128), 256, 0, stream>>>(
      hid16, w2_16, b2, x1, out, M, D_SZ, HID_SZ);
}
